// RnCLoss_26036091748816
// MI455X (gfx1250) — matrix operand order repaired and checked by reading, not compiled, not run on hardware
//
#include <hip/hip_runtime.h>
#include <hip/hip_bf16.h>
#include <math.h>

// CDNA5 / gfx1250: wave32, WMMA (not MFMA).
typedef __attribute__((ext_vector_type(16))) __bf16 v16bf;
typedef __attribute__((ext_vector_type(8)))  float  v8f;

struct alignas(16) B16x8 { unsigned int w[4]; };   // 8 bf16 = 16 bytes (global_load_b128)
union Frag { v16bf v; B16x8 q[2]; };

// ---------------------------------------------------------------------------
// labels[i] = (origin[i] - mean) / (std + 1e-8)
__global__ void k_labels(const float* __restrict__ origin, const float* __restrict__ mean,
                         const float* __restrict__ stdd, float* __restrict__ lab, int n) {
    int i = blockIdx.x * blockDim.x + threadIdx.x;
    if (i < n) lab[i] = (origin[i] - mean[0]) / (stdd[0] + 1e-8f);
}

// fp32 -> bf16 (round-to-nearest-even), packed as ushort
__global__ void k_tobf16(const float* __restrict__ f, unsigned short* __restrict__ fb, int total) {
    int i = blockIdx.x * blockDim.x + threadIdx.x;
    if (i < total) {
        unsigned int u = __float_as_uint(f[i]);
        u += 0x7FFFu + ((u >> 16) & 1u);
        fb[i] = (unsigned short)(u >> 16);
    }
}

// Bitonic sort of n (<=4096, power of 2) labels with permutation, single workgroup.
__global__ void __launch_bounds__(1024) k_sort(const float* __restrict__ lab,
                                               float* __restrict__ skey,
                                               int* __restrict__ perm, int n) {
    __shared__ float key[4096];
    __shared__ int   idx[4096];
    for (int i = threadIdx.x; i < n; i += blockDim.x) { key[i] = lab[i]; idx[i] = i; }
    __syncthreads();
    for (int k = 2; k <= n; k <<= 1) {
        for (int j = k >> 1; j > 0; j >>= 1) {
            for (int i = threadIdx.x; i < n; i += blockDim.x) {
                int ixj = i ^ j;
                if (ixj > i) {
                    bool up = ((i & k) == 0);
                    float a = key[i], b = key[ixj];
                    if ((a > b) == up) {
                        key[i] = b; key[ixj] = a;
                        int t = idx[i]; idx[i] = idx[ixj]; idx[ixj] = t;
                    }
                }
            }
            __syncthreads();
        }
    }
    for (int i = threadIdx.x; i < n; i += blockDim.x) { skey[i] = key[i]; perm[i] = idx[i]; }
}

// ---------------------------------------------------------------------------
// logits = (F * F^T) / tau  via v_wmma_f32_16x16x32_bf16.
// Block = 256 threads = 8 waves. Each wave owns one 16-row A strip (reused)
// x four 16-col B tiles. Grid: x = n/64 (cols), y = n/128 (rows).
//
// 16-bit A 16x32 layout: lane L -> m = L%16, h = L/16; halfs e=0..15 are
// K = 16h+e: one contiguous 32B chunk (same k order as B).
// 16-bit B 32x16 layout: lane L -> n = L%16, h = L/16; halfs e=0..15 are
// K = 16h+e: one contiguous 32B chunk. Since C = F*F^T, B[k][n] = F[n][k],
// so B fragments also load contiguously from F rows (no transpose needed).
__global__ void __launch_bounds__(256)
k_gemm(const unsigned short* __restrict__ fb, float* __restrict__ Lg,
       const int* __restrict__ epoch_p, int n, int d) {
    const int lane  = threadIdx.x & 31;
    const int wave  = threadIdx.x >> 5;
    const int m16   = lane & 15;
    const int h     = lane >> 4;
    const int Mbase = (blockIdx.y * 8 + wave) * 16;
    const int Nbase = blockIdx.x * 64;

    const float tau    = 1.0f - 0.9f * (float)epoch_p[0] / 100.0f;
    const float invTau = 1.0f / tau;

    const unsigned short* arow = fb + (size_t)(Mbase + m16) * d;
    const unsigned short* br0  = fb + (size_t)(Nbase +  0 + m16) * d;
    const unsigned short* br1  = fb + (size_t)(Nbase + 16 + m16) * d;
    const unsigned short* br2  = fb + (size_t)(Nbase + 32 + m16) * d;
    const unsigned short* br3  = fb + (size_t)(Nbase + 48 + m16) * d;

    v8f acc0 = {}, acc1 = {}, acc2 = {}, acc3 = {};

    for (int kk = 0; kk < d; kk += 32) {
        Frag a, b0, b1, b2, b3;
        a.q[0]  = *(const B16x8*)(arow + kk + 16 * h);
        a.q[1]  = *(const B16x8*)(arow + kk + 16 * h + 8);
        b0.q[0] = *(const B16x8*)(br0 + kk + 16 * h);
        b0.q[1] = *(const B16x8*)(br0 + kk + 16 * h + 8);
        b1.q[0] = *(const B16x8*)(br1 + kk + 16 * h);
        b1.q[1] = *(const B16x8*)(br1 + kk + 16 * h + 8);
        b2.q[0] = *(const B16x8*)(br2 + kk + 16 * h);
        b2.q[1] = *(const B16x8*)(br2 + kk + 16 * h + 8);
        b3.q[0] = *(const B16x8*)(br3 + kk + 16 * h);
        b3.q[1] = *(const B16x8*)(br3 + kk + 16 * h + 8);

        acc0 = __builtin_amdgcn_wmma_f32_16x16x32_bf16(false, a.v, false, b0.v, (short)0, acc0, false, false);
        acc1 = __builtin_amdgcn_wmma_f32_16x16x32_bf16(false, a.v, false, b1.v, (short)0, acc1, false, false);
        acc2 = __builtin_amdgcn_wmma_f32_16x16x32_bf16(false, a.v, false, b2.v, (short)0, acc2, false, false);
        acc3 = __builtin_amdgcn_wmma_f32_16x16x32_bf16(false, a.v, false, b3.v, (short)0, acc3, false, false);
    }

    // C layout: lane L -> col = L%16; VGPR r -> row = r + 8*(L/16)
#pragma unroll
    for (int r = 0; r < 8; ++r) {
        int M = Mbase + r + 8 * h;
        float* out = Lg + (size_t)M * n + Nbase;
        out[ 0 + m16] = acc0[r] * invTau;
        out[16 + m16] = acc1[r] * invTau;
        out[32 + m16] = acc2[r] * invTau;
        out[48 + m16] = acc3[r] * invTau;
    }
}

// ---------------------------------------------------------------------------
// One workgroup per sorted row position p (i = perm[p]).
// Gathers exp(logits[i, perm[r]]) into LDS in label-sorted order, builds an
// exclusive prefix sum, then each query k reduces to an interval sum found by
// two binary searches over the sorted labels:
//   denom[i,k] = total - sum_{l_j in (l_i - d, l_i + d)} exp(logit[i,j]),
//   d = |l_k - l_i|.  (diagonal j==i always falls strictly inside, matching
//   the reference's off-diagonal exclusion; ties with d_j == d_k are kept.)
__global__ void __launch_bounds__(256)
k_rows(const float* __restrict__ Lg, const float* __restrict__ skey,
       const int* __restrict__ perm, float* __restrict__ accum, int n) {
    const int p   = blockIdx.x;
    const int tid = threadIdx.x;
    const int i   = perm[p];

    __shared__ float sP[4097];    // exp row, then exclusive prefix (+ total at [n])
    __shared__ float sLab[4096];  // sorted labels
    __shared__ float chunk[256];
    __shared__ float red[256];

    const float* Lrow = Lg + (size_t)i * n;
    float logitSum = 0.f;
    for (int r = tid; r < n; r += 256) {
        float lg = Lrow[perm[r]];
        if (r != p) logitSum += lg;
        sP[r]   = __expf(lg);
        sLab[r] = skey[r];
    }
    __syncthreads();

    // per-thread chunk sums (16 each), then Hillis-Steele scan of 256 sums
    {
        int base = tid * 16;
        float s = 0.f;
#pragma unroll
        for (int e = 0; e < 16; ++e) s += sP[base + e];
        chunk[tid] = s;
    }
    __syncthreads();
    for (int off = 1; off < 256; off <<= 1) {
        float v = 0.f;
        if (tid >= off) v = chunk[tid - off];
        __syncthreads();
        if (tid >= off) chunk[tid] += v;
        __syncthreads();
    }
    {
        int base = tid * 16;
        float run = (tid == 0) ? 0.f : chunk[tid - 1];
#pragma unroll
        for (int e = 0; e < 16; ++e) { float v = sP[base + e]; sP[base + e] = run; run += v; }
        if (tid == 255) sP[n] = run;
    }
    __syncthreads();

    const float total = sP[n];
    const float sp    = sLab[p];
    float logDenomSum = 0.f;
    for (int q = tid; q < n; q += 256) {
        if (q == p) continue;
        float dd  = fabsf(sLab[q] - sp);
        float lov = sp - dd, hiv = sp + dd;
        // a = first r with sLab[r] > lov  (upper_bound)
        int a = 0;
        { int len = n;
          while (len > 0) { int half = len >> 1; int mid = a + half;
              if (sLab[mid] <= lov) { a = mid + 1; len -= half + 1; } else len = half; } }
        // b = first r with sLab[r] >= hiv (lower_bound)
        int b = 0;
        { int len = n;
          while (len > 0) { int half = len >> 1; int mid = b + half;
              if (sLab[mid] < hiv) { b = mid + 1; len -= half + 1; } else len = half; } }
        float denom = total - (sP[b] - sP[a]);
        logDenomSum += __logf(denom);
    }

    // block reductions + two atomics
    red[tid] = logitSum; __syncthreads();
    for (int s = 128; s > 0; s >>= 1) { if (tid < s) red[tid] += red[tid + s]; __syncthreads(); }
    float blockLogit = red[0];
    __syncthreads();
    red[tid] = logDenomSum; __syncthreads();
    for (int s = 128; s > 0; s >>= 1) { if (tid < s) red[tid] += red[tid + s]; __syncthreads(); }
    if (tid == 0) {
        atomicAdd(&accum[0], blockLogit);
        atomicAdd(&accum[1], red[0]);
    }
}

__global__ void k_zero(float* accum) { if (threadIdx.x < 2) accum[threadIdx.x] = 0.f; }

__global__ void k_final(const float* __restrict__ accum, float* __restrict__ out, int n) {
    out[0] = -(accum[0] - accum[1]) / ((float)n * (float)(n - 1));
}

// ---------------------------------------------------------------------------
extern "C" void kernel_launch(void* const* d_in, const int* in_sizes, int n_in,
                              void* d_out, int out_size, void* d_ws, size_t ws_size,
                              hipStream_t stream) {
    const float* features = (const float*)d_in[0];
    const float* origin   = (const float*)d_in[1];
    const float* mean     = (const float*)d_in[2];
    const float* stdd     = (const float*)d_in[3];
    const int*   epoch    = (const int*)d_in[4];

    const int n = in_sizes[1];            // origin_label: (N,1) -> N
    const int d = in_sizes[0] / n;        // features: (N,D)
    float* out = (float*)d_out;

    // workspace layout (all 16B-aligned by construction): ~69.3 MB total
    char* ws = (char*)d_ws;
    size_t off = 0;
    float*          Lg   = (float*)(ws + off);          off += (size_t)n * n * sizeof(float);
    unsigned short* fb   = (unsigned short*)(ws + off); off += (size_t)n * d * sizeof(unsigned short);
    float*          lab  = (float*)(ws + off);          off += (size_t)n * sizeof(float);
    float*          skey = (float*)(ws + off);          off += (size_t)n * sizeof(float);
    int*            perm = (int*)(ws + off);            off += (size_t)n * sizeof(int);
    float*          acc  = (float*)(ws + off);          off += 2 * sizeof(float);
    (void)ws_size; (void)n_in; (void)out_size;

    k_zero  <<<1, 32, 0, stream>>>(acc);
    k_labels<<<(n + 255) / 256, 256, 0, stream>>>(origin, mean, stdd, lab, n);
    k_tobf16<<<((n * d) + 255) / 256, 256, 0, stream>>>(features, fb, n * d);
    k_sort  <<<1, 1024, 0, stream>>>(lab, skey, perm, n);

    dim3 gg(n / 64, n / 128);             // 64 x 32 blocks, 8 waves each
    k_gemm  <<<gg, 256, 0, stream>>>(fb, Lg, epoch, n, d);

    k_rows  <<<n, 256, 0, stream>>>(Lg, skey, perm, acc, n);
    k_final <<<1, 1, 0, stream>>>(acc, out, n);
}
